// MyModel_87522843559628
// MI455X (gfx1250) — matrix operand order repaired and checked by reading, not compiled, not run on hardware
//
#include <hip/hip_runtime.h>
#include <stdint.h>

// ---------------------------------------------------------------------------
// Persistent stacked-LSTM for MI455X (gfx1250, wave32, WMMA bf16 -> f32 acc).
//
// B=64, T=512, D=H=512, L=3.  Per layer-step GEMM: [64 x 1024] @ [1024 x 2048].
// - Weights converted once to bf16, pre-transposed Bt[l][n][k]; each WG pins
//   its 128-column slice (258 KB) in LDS for the whole sequence.
// - x converted once to bf16 in [T,B,D] layout so every layer stages its A
//   tile with the same contiguous bf16 copy; A staging is register
//   double-buffered (4xuint4/thread, 128-col chunks) so L2 latency hides
//   under the previous chunk's WMMAs without exceeding the 256-VGPR window.
// - 3 layers x 16 hidden-slices = 48 persistent WGs form a software pipeline
//   synchronized with per-(layer,t) atomic counters (acquire/release, agent).
// ---------------------------------------------------------------------------

typedef __attribute__((ext_vector_type(16))) __bf16 v16bf;
typedef __attribute__((ext_vector_type(8)))  float  v8f;

#define T_STEPS 512
#define RING    8
#define NSLICE  16
#define LAYERS  3

// LDS geometry (bytes)
#define B_ROW_STRIDE  2064                   // 1024 bf16 + 8 pad halves
#define B_LDS_BYTES   (128 * B_ROW_STRIDE)   // 264192
#define A_ROW_STRIDE  272                    // 128 bf16 + 8 pad halves
#define Z_ROW_STRIDE  528                    // 128 f32 + 4 pad floats
#define AZ_LDS_BYTES  (64 * Z_ROW_STRIDE)    // 33792 (A region is a subset)

struct Pack32 { uint4 a, b; };

__device__ __forceinline__ v16bf ld_frag(const unsigned char* p0,
                                         const unsigned char* p1) {
  Pack32 s;
  s.a = *(const uint4*)p0;
  s.b = *(const uint4*)p1;
  return __builtin_bit_cast(v16bf, s);
}

__device__ __forceinline__ unsigned short f2bf(float f) {
  union { float f; unsigned u; } v; v.f = f;
  unsigned r = v.u + 0x7FFFu + ((v.u >> 16) & 1u);   // round-to-nearest-even
  return (unsigned short)(r >> 16);
}

__device__ __forceinline__ void spin_wait(int* p, int target) {
  while (__hip_atomic_load(p, __ATOMIC_ACQUIRE, __HIP_MEMORY_SCOPE_AGENT) < target)
    __builtin_amdgcn_s_sleep(2);
}

__device__ __forceinline__ float sigmoidf_fast(float x) {
  return 1.0f / (1.0f + __expf(-x));
}

// --------------------------- one-shot conversions --------------------------
// Wt[l][n][k] (bf16) = k<512 ? W[l][k][n] : U[l][k-512][n]
__global__ __launch_bounds__(256) void cvt_weights_kernel(
    const float* __restrict__ W, const float* __restrict__ U,
    unsigned short* __restrict__ Wt) {
  size_t lin = (size_t)blockIdx.x * 256 + threadIdx.x;  // < 3*2048*1024
  int k = (int)(lin & 1023);
  int n = (int)((lin >> 10) & 2047);
  int l = (int)(lin >> 21);
  float v = (k < 512) ? W[((size_t)(l * 512 + k)) * 2048 + n]
                      : U[((size_t)(l * 512 + (k - 512))) * 2048 + n];
  Wt[lin] = f2bf(v);
}

// xbf[t][b][d] (bf16) = x[b][t][d]  -- contiguous 64x512 tile per timestep
__global__ __launch_bounds__(256) void cvt_x_kernel(
    const float* __restrict__ x, unsigned short* __restrict__ xbf) {
  size_t lin = (size_t)blockIdx.x * 256 + threadIdx.x;  // < 512*64*512
  int d = (int)(lin & 511);
  int b = (int)((lin >> 9) & 63);
  int t = (int)(lin >> 15);
  xbf[lin] = f2bf(x[((size_t)b * 512 + t) * 512 + d]);
}

__global__ __launch_bounds__(256) void zero_done_kernel(int* __restrict__ done) {
  for (int i = threadIdx.x; i < LAYERS * T_STEPS; i += 256) done[i] = 0;
}

// ----------------------------- main kernel ---------------------------------
__global__ __launch_bounds__(256, 2) void lstm_persistent_kernel(
    const unsigned short* __restrict__ xbf, // [512,64,512] bf16 bits
    const float* __restrict__ bias,         // [3,2048]     f32
    const unsigned short* __restrict__ Wt,  // [3,2048,1024] bf16 bits
    unsigned short* __restrict__ hbuf,      // [3,RING,64,512] bf16 bits
    int* __restrict__ done,                 // [3,512]
    float* __restrict__ out)                // [64,512] f32
{
  __shared__ __align__(16) unsigned char sB[B_LDS_BYTES];
  __shared__ __align__(16) unsigned char sAZ[AZ_LDS_BYTES];

  const int l   = blockIdx.x >> 4;     // layer 0..2
  const int hs  = blockIdx.x & 15;     // hidden slice 0..15
  const int tid = threadIdx.x;

  // ---- pin weight slice in LDS (once) -------------------------------------
  {
    int n_local = tid >> 1;                       // 0..127
    int g  = n_local >> 5;                        // gate 0..3 (i,f,g,o)
    int jj = n_local & 31;
    int nglob = g * 512 + hs * 32 + jj;
    const uint4* src = (const uint4*)(Wt + ((size_t)(l * 2048 + nglob)) * 1024)
                       + (tid & 1) * 64;
    uint4* dst = (uint4*)(sB + n_local * B_ROW_STRIDE + (tid & 1) * 1024);
#pragma unroll 4
    for (int i = 0; i < 64; ++i) dst[i] = src[i];
  }

  // ---- per-thread gate-math constants / state -----------------------------
  const int jl    = tid & 31;
  const int jglob = hs * 32 + jl;
  const float* bl = bias + (size_t)l * 2048;
  const float bi = bl[jglob], bff = bl[512 + jglob],
              bg = bl[1024 + jglob], bo = bl[1536 + jglob];
  const int r0 = (tid >> 5) * 8;
  float cst[8];
#pragma unroll
  for (int k = 0; k < 8; ++k) cst[k] = 0.0f;

  // ---- per-wave WMMA geometry ---------------------------------------------
  const int lane = tid & 31;
  const int wave = tid >> 5;
  const int mt   = wave >> 1;                 // M tile 0..3
  const int nt0  = (wave & 1) * 4;            // first of 4 N tiles
  const int hiHalf = lane >> 4;
  const int mrow   = mt * 16 + (lane & 15);
  const int a_row_byte = mrow * A_ROW_STRIDE;
  const int kb_a = hiHalf * 8;
  const int kb_b = hiHalf * 8;

  // ---- A-staging geometry (register double-buffer, 128-col chunks) --------
  const int st_r  = tid >> 2;                 // row 0..63
  const int st_c0 = (tid & 3) * 32;           // 32 bf16 columns per thread
  uint4* const st_dst = (uint4*)(sAZ + (size_t)st_r * A_ROW_STRIDE + st_c0 * 2);

  const v8f vzero = {0.f, 0.f, 0.f, 0.f, 0.f, 0.f, 0.f, 0.f};

  __syncthreads();

  // =========================== time loop ===================================
  for (int t = 0; t < T_STEPS; ++t) {
    if (tid == 0) {
      if (t > 0) spin_wait(&done[l * T_STEPS + (t - 1)], NSLICE);        // recurrence
      if (l > 0) spin_wait(&done[(l - 1) * T_STEPS + t], NSLICE);        // input
      if (l < LAYERS - 1 && t >= RING)
        spin_wait(&done[(l + 1) * T_STEPS + (t - RING)], NSLICE);        // ring b/p
    }
    __syncthreads();

    // Source bases for this step's A = [input(512) | recurrent(512)], bf16.
    const unsigned short* inpbase =
        (l == 0) ? (xbf + (size_t)t * 64 * 512)
                 : (hbuf + (((size_t)(l - 1) * RING + (t & (RING - 1))) * 64) * 512);
    const unsigned short* recbase =
        hbuf + (((size_t)l * RING + ((t - 1) & (RING - 1))) * 64) * 512;

    uint4 stg[4];
    // prologue: issue loads for chunk 0 (always from input half)
    {
      const uint4* src = (const uint4*)(inpbase + (size_t)st_r * 512 + st_c0);
#pragma unroll
      for (int i = 0; i < 4; ++i) stg[i] = src[i];
    }

    v8f acc[4];
#pragma unroll
    for (int n = 0; n < 4; ++n) acc[n] = vzero;

    // ---- GEMM over K = 1024 in 8 chunks of 128 ----------------------------
    for (int kc = 0; kc < 8; ++kc) {
      __syncthreads();   // A region free (prev wmma reads / prev z reads done)

      // store previously loaded chunk into LDS
#pragma unroll
      for (int i = 0; i < 4; ++i) st_dst[i] = stg[i];

      // issue loads for next chunk (overlaps with this chunk's WMMAs)
      if (kc < 7) {
        int gk = (kc + 1) * 128 + st_c0;
        if (gk < 512) {
          const uint4* src = (const uint4*)(inpbase + (size_t)st_r * 512 + gk);
#pragma unroll
          for (int i = 0; i < 4; ++i) stg[i] = src[i];
        } else if (t == 0) {
          uint4 z4 = {0u, 0u, 0u, 0u};
#pragma unroll
          for (int i = 0; i < 4; ++i) stg[i] = z4;
        } else {
          const uint4* src =
              (const uint4*)(recbase + (size_t)st_r * 512 + (gk - 512));
#pragma unroll
          for (int i = 0; i < 4; ++i) stg[i] = src[i];
        }
      }
      __syncthreads();   // staged chunk visible to all waves

      // 4 K-steps of WMMA over this chunk
#pragma unroll
      for (int kk = 0; kk < 4; ++kk) {
        const int ka = kk * 32;                        // chunk-local K (halves)
        v16bf af = ld_frag(sAZ + a_row_byte + (ka + kb_a) * 2,
                           sAZ + a_row_byte + (ka + kb_a + 16) * 2);
        const int kbyte = (kc * 128 + ka + kb_b) * 2;  // global K offset in sB
#pragma unroll
        for (int n = 0; n < 4; ++n) {
          const unsigned char* bp =
              sB + ((nt0 + n) * 16 + (lane & 15)) * B_ROW_STRIDE + kbyte;
          v16bf bfrag = ld_frag(bp, bp + 32);
          acc[n] = __builtin_amdgcn_wmma_f32_16x16x32_bf16(
              false, af, false, bfrag, (short)0, acc[n], false, false);
        }
      }
    }
    __syncthreads();   // all waves done reading A region -> reuse as z

    // ---- scatter accumulators (z) to LDS ----------------------------------
#pragma unroll
    for (int n = 0; n < 4; ++n) {
#pragma unroll
      for (int v = 0; v < 8; ++v) {
        int zrow = mt * 16 + v + (hiHalf << 3);
        int zcol = (nt0 + n) * 16 + (lane & 15);
        *(float*)(sAZ + (size_t)zrow * Z_ROW_STRIDE + zcol * 4) = acc[n][v];
      }
    }
    __syncthreads();

    // ---- gate math, c/h update --------------------------------------------
#pragma unroll
    for (int k = 0; k < 8; ++k) {
      int r = r0 + k;
      const float* zr = (const float*)(sAZ + (size_t)r * Z_ROW_STRIDE);
      float zi = zr[jl]      + bi;
      float zf = zr[32 + jl] + bff;
      float zg = zr[64 + jl] + bg;
      float zo = zr[96 + jl] + bo;
      float ig = sigmoidf_fast(zi);
      float fg = sigmoidf_fast(zf);
      float gg = tanhf(zg);
      float og = sigmoidf_fast(zo);
      float cn = fg * cst[k] + ig * gg;
      cst[k] = cn;
      float hn = og * tanhf(cn);
      hbuf[(((size_t)l * RING + (t & (RING - 1))) * 64 + r) * 512 + jglob] =
          f2bf(hn);
      if (l == LAYERS - 1 && t == T_STEPS - 1) out[(size_t)r * 512 + jglob] = hn;
    }

    // prefetch next timestep's x tile (layer-0 WGs; 256 threads x 256 B = 64 KB)
    if (l == 0 && t + 1 < T_STEPS)
      __builtin_prefetch(xbf + (size_t)(t + 1) * 64 * 512 + tid * 128, 0, 3);

    __threadfence();     // make this thread's hbuf stores agent-visible
    __syncthreads();     // all threads' stores done before the signal
    if (tid == 0)
      __hip_atomic_fetch_add(&done[l * T_STEPS + t], 1,
                             __ATOMIC_RELEASE, __HIP_MEMORY_SCOPE_AGENT);
  }
}

// ------------------------------ launcher -----------------------------------
extern "C" void kernel_launch(void* const* d_in, const int* in_sizes, int n_in,
                              void* d_out, int out_size, void* d_ws, size_t ws_size,
                              hipStream_t stream) {
  const float* x = (const float*)d_in[0];   // [64,512,512]
  const float* W = (const float*)d_in[1];   // [3,512,2048]
  const float* U = (const float*)d_in[2];   // [3,512,2048]
  const float* b = (const float*)d_in[3];   // [3,2048]
  float* out = (float*)d_out;               // [64,512]

  unsigned char* ws = (unsigned char*)d_ws;
  int* done            = (int*)ws;                                    // 8 KB
  unsigned short* hbuf = (unsigned short*)(ws + 8192);                // 1.5 MB
  unsigned short* Wt   = (unsigned short*)(ws + 8192 + 1572864);      // 12.6 MB
  unsigned short* xbf  = (unsigned short*)(ws + 8192 + 1572864 + 12582912); // 32 MB

  zero_done_kernel<<<1, 256, 0, stream>>>(done);
  cvt_weights_kernel<<<(LAYERS * 2048 * 1024) / 256, 256, 0, stream>>>(W, U, Wt);
  cvt_x_kernel<<<(512 * 64 * 512) / 256, 256, 0, stream>>>(x, xbf);
  lstm_persistent_kernel<<<LAYERS * NSLICE, 256, 0, stream>>>(xbf, b, Wt, hbuf,
                                                              done, out);
}
